// FP8ScaledLayer_24060406792516
// MI455X (gfx1250) — matrix operand order repaired and checked by reading, not compiled, not run on hardware
//
#include <hip/hip_runtime.h>

// ---- CDNA5 (gfx1250) vector types -----------------------------------------
typedef __attribute__((ext_vector_type(16))) _Float16 v16h;
typedef __attribute__((ext_vector_type(8)))  _Float16 v8h;
typedef __attribute__((ext_vector_type(4)))  _Float16 v4h;
typedef __attribute__((ext_vector_type(8)))  float    v8f;
typedef __attribute__((ext_vector_type(4)))  float    v4f;
typedef __attribute__((__vector_size__(16))) int      v4i;   // async builtin pointee

#define GAS __attribute__((address_space(1)))
#define LAS __attribute__((address_space(3)))

#if defined(__has_builtin)
#if __has_builtin(__builtin_amdgcn_global_load_async_to_lds_b128)
#define HAVE_ASYNC_LDS 1
#endif
#if __has_builtin(__builtin_amdgcn_s_wait_asynccnt)
#define HAVE_WAIT_ASYNC_BUILTIN 1
#endif
#endif

// ---- problem sizes (B=4, S=4096, K=4096, N=4096) --------------------------
static constexpr int MTOT = 16384;   // B*S rows
static constexpr int KTOT = 4096;
static constexpr int NTOT = 4096;

// ---- tiling ----------------------------------------------------------------
// Block: 128(M) x 256(N), 8 waves as 2(M) x 4(N), each wave 64x64 = 4x4 WMMA
// tiles -> 16 wmma per 16 ds_load_b128 per k-step (square register tile).
static constexpr int BM = 128;
static constexpr int BN = 256;
static constexpr int BK = 64;        // K tile per LDS stage (2 WMMA k-steps)
static constexpr int LD = BK + 8;    // LDS row stride in halves (144 B)

__device__ inline void wait_async_lds() {
#if defined(HAVE_WAIT_ASYNC_BUILTIN)
  __builtin_amdgcn_s_wait_asynccnt(0);
#elif defined(HAVE_ASYNC_LDS)
  asm volatile("s_wait_asynccnt 0" ::: "memory");
#endif
}

__global__ __launch_bounds__(256) void fp8scaled_linear_wmma(
    const float* __restrict__ x,        // [MTOT, KTOT] fp32
    const _Float16* __restrict__ w,     // [NTOT, KTOT] fp16
    const float* __restrict__ scale,    // [NTOT]
    const float* __restrict__ bias,     // [NTOT]
    float* __restrict__ out)            // [MTOT, NTOT]
{
  __shared__ __align__(16) _Float16 As[2][BM * LD];  // x tile, fp16-converted
  __shared__ __align__(16) _Float16 Bs[2][BN * LD];  // weight tile

  const int t    = threadIdx.x;
  const int wave = t >> 5;
  const int lane = t & 31;
  const int hlf  = lane >> 4;   // 0: lanes 0-15, 1: lanes 16-31
  const int l16  = lane & 15;
  const int wn   = wave & 3;    // wave N index (4 waves along N, 64 cols each)
  const int wm   = wave >> 2;   // wave M index (2 waves along M, 64 rows each)

  const int gn0 = blockIdx.x * BN;
  const int gm0 = blockIdx.y * BM;

  v8f acc[4][4] = {};           // 4 M-subtiles x 4 N-subtiles of 16x16 f32

  // ---- stage loader: global -> LDS for K-slice kt -------------------------
  auto load_stage = [&](int kt, int buf) {
    const int k0 = kt * BK;
    // weight tile: 256 rows x 64 halves = 2048 x 16B chunks; 8 per thread.
    // async global->LDS copies (ASYNCcnt) on gfx1250.
#pragma unroll
    for (int i = 0; i < 8; ++i) {
      const int f   = t + 256 * i;
      const int row = f >> 3;       // 0..255
      const int c8  = f & 7;        // 8-half chunk in row
      const _Float16* gsrc = w + (size_t)(gn0 + row) * KTOT + k0 + c8 * 8;
      _Float16* ldst = &Bs[buf][row * LD + c8 * 8];
#if defined(HAVE_ASYNC_LDS)
      __builtin_amdgcn_global_load_async_to_lds_b128(
          (GAS v4i*)gsrc, (LAS v4i*)ldst, 0, 0);
#else
      *(v8h*)ldst = *(const v8h*)gsrc;
#endif
    }
    // x tile: 128 rows x 64 fp32; global_load_b128, cvt_pk_f16_f32, ds_store.
#pragma unroll
    for (int i = 0; i < 8; ++i) {
      const int f   = t + 256 * i;
      const int row = f >> 4;       // 0..127
      const int c4  = f & 15;       // float4 within row
      const v4f xv = *(const v4f*)(x + (size_t)(gm0 + row) * KTOT + k0 + c4 * 4);
      v4h hv;
      hv[0] = (_Float16)xv[0];
      hv[1] = (_Float16)xv[1];
      hv[2] = (_Float16)xv[2];
      hv[3] = (_Float16)xv[3];
      *(v4h*)(&As[buf][row * LD + c4 * 4]) = hv;
    }
  };

  // ---- compute one LDS stage (2 WMMA k-steps of 32) -----------------------
  auto compute_stage = [&](int buf) {
#pragma unroll
    for (int ks = 0; ks < 2; ++ks) {
      v16h a[4], b[4];
      // A fragments (16x32 f16): lane hlf selects K base 0 or 8; second chunk
      // at +16 halves covers K=kb+16..kb+23 per ISA VGPR map.
#pragma unroll
      for (int mi = 0; mi < 4; ++mi) {
        const int m = wm * 64 + mi * 16 + l16;
        const _Float16* p = &As[buf][m * LD + ks * 32 + hlf * 8];
        union { v16h v; v8h h[2]; } u;
        u.h[0] = *(const v8h*)p;          // K = kb .. kb+7
        u.h[1] = *(const v8h*)(p + 16);   // K = kb+16 .. kb+23
        a[mi] = u.v;
      }
      // B fragments (32x16 f16): same K order as A (base 0 or 8, second chunk +16).
#pragma unroll
      for (int ni = 0; ni < 4; ++ni) {
        const int n = wn * 64 + ni * 16 + l16;
        const _Float16* p = &Bs[buf][n * LD + ks * 32 + hlf * 8];
        union { v16h v; v8h h[2]; } u;
        u.h[0] = *(const v8h*)p;          // K = kb .. kb+7
        u.h[1] = *(const v8h*)(p + 16);   // K = kb+16 .. kb+23
        b[ni] = u.v;
      }
#pragma unroll
      for (int mi = 0; mi < 4; ++mi)
#pragma unroll
        for (int ni = 0; ni < 4; ++ni)
          acc[mi][ni] = __builtin_amdgcn_wmma_f32_16x16x32_f16(
              false, a[mi], false, b[ni], (short)0, acc[mi][ni], false, false);
    }
  };

  // ---- main loop: double-buffered pipeline --------------------------------
  load_stage(0, 0);
  const int KT = KTOT / BK;  // 64 stages
  for (int kt = 0; kt < KT; ++kt) {
    const int buf = kt & 1;
    wait_async_lds();          // drain this wave's async B-tile copies
    __syncthreads();           // all waves done with buf^1 reads + buf writes
    if (kt + 1 < KT) load_stage(kt + 1, buf ^ 1);
    compute_stage(buf);
  }

  // ---- epilogue: per-out-feature scale + bias, f32 store ------------------
  // C/D layout: VGPR r of lane L holds (M = r + 8*(L/16), N = L%16).
#pragma unroll
  for (int ni = 0; ni < 4; ++ni) {
    const int gn = gn0 + wn * 64 + ni * 16 + l16;
    const float s  = scale[gn];
    const float bz = bias[gn];
#pragma unroll
    for (int mi = 0; mi < 4; ++mi) {
      const int gmb = gm0 + wm * 64 + mi * 16 + hlf * 8;
#pragma unroll
      for (int r = 0; r < 8; ++r) {
        out[(size_t)(gmb + r) * NTOT + gn] = acc[mi][ni][r] * s + bz;
      }
    }
  }
}

extern "C" void kernel_launch(void* const* d_in, const int* in_sizes, int n_in,
                              void* d_out, int out_size, void* d_ws, size_t ws_size,
                              hipStream_t stream) {
  (void)in_sizes; (void)n_in; (void)out_size; (void)d_ws; (void)ws_size;
  const float*    x  = (const float*)d_in[0];
  const _Float16* w  = (const _Float16*)d_in[1];   // fp16 weight storage
  const float*    sc = (const float*)d_in[2];
  const float*    bz = (const float*)d_in[3];
  float*          o  = (float*)d_out;

  dim3 grid(NTOT / BN, MTOT / BM);   // 16 x 128 blocks, N fastest (A-tile reuse)
  fp8scaled_linear_wmma<<<grid, dim3(256), 0, stream>>>(x, w, sc, bz, o);
}
